// EGNN_4380866641945
// MI455X (gfx1250) — matrix operand order repaired and checked by reading, not compiled, not run on hardware
//
#include <hip/hip_runtime.h>

typedef _Float16 f16;
typedef __attribute__((ext_vector_type(16))) _Float16 v16h;
typedef __attribute__((ext_vector_type(8)))  float    v8f;

#define NN 50000
#define NE 800000
#define NG 512
#define HID 128
#define NL 6
#define EW 4   // waves per block in edge kernel

union AB { v16h v; uint4 u[2]; };

__device__ __forceinline__ v8f wmma16(v16h a, v16h b, v8f c) {
    // v_wmma_f32_16x16x32_f16: D = A(16x32) x B(32x16) + C(16x16 f32)
    return __builtin_amdgcn_wmma_f32_16x16x32_f16(false, a, false, b, (short)0, c, false, false);
}

__device__ __forceinline__ v8f wmma16_z(v16h a, v16h b) {
    // start accumulation chain from inline-0 C
    v8f z = {};
    return __builtin_amdgcn_wmma_f32_16x16x32_f16(false, a, false, b, (short)0, z, false, false);
}

__device__ __forceinline__ float sspf(float x) {
    // softplus(x) - 0.5, numerically stable
    return fmaxf(x, 0.0f) + log1pf(expf(-fabsf(x))) - 0.5f;
}

// ---- pack a [K=128][N=128] row-major f32 weight into wave-native WMMA B layout ----
// packed[((t*4+ks)*32+lane)*16+e] = W[(ks*32+(lane>>4)*8+(e&7)+(e>>3)*16)*128 + t*16+(lane&15)]
__global__ void k_pack128(const float* __restrict__ W, f16* __restrict__ out) {
    int p = blockIdx.x * 256 + threadIdx.x;           // 16384 threads
    int e = p & 15, lane = (p >> 4) & 31, ks = (p >> 9) & 3, t = p >> 11;
    int col = t * 16 + (lane & 15);
    int K = ks * 32 + (lane >> 4) * 8 + (e & 7) + (e >> 3) * 16;
    out[p] = (f16)W[K * HID + col];
}

// ---- pack a [K=32][N=128] f32 weight (fw1) ----
__global__ void k_pack32(const float* __restrict__ W, f16* __restrict__ out) {
    int p = blockIdx.x * 256 + threadIdx.x;           // 4096 threads
    int e = p & 15, lane = (p >> 4) & 31, t = p >> 9;
    int col = t * 16 + (lane & 15);
    int K = (lane >> 4) * 8 + (e & 7) + (e >> 3) * 16;
    out[p] = (f16)W[K * HID + col];
}

// ---- rbf (f16) + degree ----
__global__ void k_pre(const float* __restrict__ pos, const int* __restrict__ ei,
                      f16* __restrict__ rbf, float* __restrict__ deg) {
    int e = blockIdx.x * 256 + threadIdx.x;
    if (e >= NE) return;
    int i = ei[e], j = ei[NE + e];
    float dx = pos[i*3+0] - pos[j*3+0];
    float dy = pos[i*3+1] - pos[j*3+1];
    float dz = pos[i*3+2] - pos[j*3+2];
    float d = sqrtf(dx*dx + dy*dy + dz*dz);
    #pragma unroll
    for (int k = 0; k < 32; k++) {
        float c = 4.0f * (float)k / 31.0f;
        float t = d - c;
        rbf[(size_t)e * 32 + k] = (f16)expf(-32.0f * t * t);
    }
    atomicAdd(&deg[i], 1.0f);
}

__global__ void k_invdeg(float* __restrict__ d) {
    int n = blockIdx.x * 256 + threadIdx.x;
    if (n < NN) d[n] = 1.0f / fmaxf(d[n], 1.0f);
}

__global__ void k_inith(const int* __restrict__ z, const float* __restrict__ embed,
                        float* __restrict__ h) {
    int t = blockIdx.x * 256 + threadIdx.x;           // NN*HID exact
    int n = t >> 7, c = t & 127;
    h[t] = embed[z[n] * HID + c];
}

// ---- edge kernel: W = ssp(rbf@fw1+b1)@fw2+b2 ; scatter h[j]*W into aggr[i] ----
// Weights async-staged into LDS once per block (ASYNCcnt path), B operands fed from LDS.
__global__ __launch_bounds__(EW * 32) void k_edge(
    const f16* __restrict__ rbf, const int* __restrict__ ei,
    const float* __restrict__ h, float* __restrict__ aggr,
    const f16* __restrict__ fw1p, const float* __restrict__ fb1,
    const f16* __restrict__ fw2p, const float* __restrict__ fb2) {
    __shared__ __align__(16) f16 smem[4096 + 16384 + EW * 16 * 136];
    f16* wb1 = smem;                    // 8 KB  packed fw1 (B layout)
    f16* wb2 = smem + 4096;             // 32 KB packed fw2 (B layout)
    f16* stg = smem + 4096 + 16384;     // per-wave ssp staging

    // ---- async stage packed weights into LDS (global_load_async_to_lds_b128) ----
    {
        unsigned lds1 = (unsigned)(size_t)&wb1[0];   // flat->LDS offset: addr[31:0]
        unsigned lds2 = (unsigned)(size_t)&wb2[0];
        for (int c = threadIdx.x; c < 512; c += EW * 32) {     // 8 KB
            unsigned lo = lds1 + c * 16;
            unsigned go = c * 16;
            asm volatile("global_load_async_to_lds_b128 %0, %1, %2"
                         :: "v"(lo), "v"(go), "s"(fw1p) : "memory");
        }
        for (int c = threadIdx.x; c < 2048; c += EW * 32) {    // 32 KB
            unsigned lo = lds2 + c * 16;
            unsigned go = c * 16;
            asm volatile("global_load_async_to_lds_b128 %0, %1, %2"
                         :: "v"(lo), "v"(go), "s"(fw2p) : "memory");
        }
        asm volatile("s_wait_asynccnt 0" ::: "memory");
    }
    __syncthreads();

    const int lane = threadIdx.x & 31;
    const int wv = threadIdx.x >> 5;
    const int tile = blockIdx.x * EW + wv;            // 50000 tiles exact
    const int m = lane & 15, kh = lane >> 4;
    const int e0 = tile * 16;
    f16* w = &stg[wv * 16 * 136];

    // prefetch next block's rbf rows for this lane (global_prefetch_b8)
    if (e0 + EW * 16 + m < NE)
        __builtin_prefetch((const void*)(rbf + (size_t)(e0 + EW * 16 + m) * 32), 0, 3);

    // A = rbf tile 16x32 (ISA 16-bit A layout: two 16B runs per lane)
    AB a;
    const uint4* rp = (const uint4*)(rbf + (size_t)(e0 + m) * 32);
    a.u[0] = rp[kh];
    a.u[1] = rp[2 + kh];

    // GEMM1: zero-C chain, bias fused into ssp epilogue
    v8f acc[8];
    #pragma unroll
    for (int t = 0; t < 8; t++) {
        AB b;
        const uint4* bp = (const uint4*)&wb1[(t * 32 + lane) * 16];
        b.u[0] = bp[0]; b.u[1] = bp[1];
        acc[t] = wmma16_z(a.v, b.v);
    }
    // ssp(x + b1) -> LDS (f16, C layout: row = r + kh*8, col = t*16+m)
    #pragma unroll
    for (int t = 0; t < 8; t++) {
        int col = t * 16 + m;
        float bv = fb1[col];
        #pragma unroll
        for (int r = 0; r < 8; r++)
            w[(r + kh * 8) * 136 + col] = (f16)sspf(acc[t][r] + bv);
    }
    __syncthreads();

    // GEMM2: zero-C chain (peel ks=0), bias fused into message epilogue
    v8f acc2[8];
    #pragma unroll
    for (int ks = 0; ks < 4; ks++) {
        AB a2;
        a2.u[0] = *(const uint4*)&w[m * 136 + ks * 32 + kh * 8];
        a2.u[1] = *(const uint4*)&w[m * 136 + ks * 32 + 16 + kh * 8];
        #pragma unroll
        for (int t = 0; t < 8; t++) {
            AB b;
            const uint4* bp = (const uint4*)&wb2[((t * 4 + ks) * 32 + lane) * 16];
            b.u[0] = bp[0]; b.u[1] = bp[1];
            acc2[t] = (ks == 0) ? wmma16_z(a2.v, b.v) : wmma16(a2.v, b.v, acc2[t]);
        }
    }
    int vi[8], vj[8];
    #pragma unroll
    for (int r = 0; r < 8; r++) {
        int e = e0 + r + kh * 8;
        vi[r] = ei[e]; vj[r] = ei[NE + e];
    }
    #pragma unroll
    for (int t = 0; t < 8; t++) {
        int col = t * 16 + m;
        float bv = fb2[col];
        #pragma unroll
        for (int r = 0; r < 8; r++) {
            float msg = h[(size_t)vj[r] * HID + col] * (acc2[t][r] + bv);
            atomicAdd(&aggr[(size_t)vi[r] * HID + col], msg);
        }
    }
}

// ---- node update: h = ssp((aggr*invdeg)@uw1+b1)@uw2+b2 ----
__global__ __launch_bounds__(32) void k_node(
    const float* __restrict__ aggr, const float* __restrict__ invd, float* __restrict__ h,
    const f16* __restrict__ uw1p, const float* __restrict__ ub1,
    const f16* __restrict__ uw2p, const float* __restrict__ ub2) {
    __shared__ __align__(16) f16 w[16 * 136];
    const int lane = threadIdx.x;
    const int m = lane & 15, kh = lane >> 4;
    const int n0 = blockIdx.x * 16;                   // 3125 blocks exact
    const float sc = invd[n0 + m];
    const float* src = aggr + (size_t)(n0 + m) * HID;

    v8f acc[8];
    #pragma unroll
    for (int ks = 0; ks < 4; ks++) {
        v16h av;
        #pragma unroll
        for (int q = 0; q < 8; q++) {
            av[q]     = (f16)(src[ks * 32 + kh * 8 + q] * sc);
            av[q + 8] = (f16)(src[ks * 32 + 16 + kh * 8 + q] * sc);
        }
        #pragma unroll
        for (int t = 0; t < 8; t++) {
            AB b;
            const uint4* bp = (const uint4*)(uw1p + (size_t)((t * 4 + ks) * 32 + lane) * 16);
            b.u[0] = bp[0]; b.u[1] = bp[1];
            acc[t] = (ks == 0) ? wmma16_z(av, b.v) : wmma16(av, b.v, acc[t]);
        }
    }
    #pragma unroll
    for (int t = 0; t < 8; t++) {
        int col = t * 16 + m;
        float bv = ub1[col];
        #pragma unroll
        for (int r = 0; r < 8; r++)
            w[(r + kh * 8) * 136 + col] = (f16)sspf(acc[t][r] + bv);
    }
    __syncthreads();
    v8f acc2[8];
    #pragma unroll
    for (int ks = 0; ks < 4; ks++) {
        AB a2;
        a2.u[0] = *(const uint4*)&w[m * 136 + ks * 32 + kh * 8];
        a2.u[1] = *(const uint4*)&w[m * 136 + ks * 32 + 16 + kh * 8];
        #pragma unroll
        for (int t = 0; t < 8; t++) {
            AB b;
            const uint4* bp = (const uint4*)(uw2p + (size_t)((t * 4 + ks) * 32 + lane) * 16);
            b.u[0] = bp[0]; b.u[1] = bp[1];
            acc2[t] = (ks == 0) ? wmma16_z(a2.v, b.v) : wmma16(a2.v, b.v, acc2[t]);
        }
    }
    #pragma unroll
    for (int t = 0; t < 8; t++) {
        int col = t * 16 + m;
        float bv = ub2[col];
        #pragma unroll
        for (int r = 0; r < 8; r++)
            h[(size_t)(n0 + r + kh * 8) * HID + col] = acc2[t][r] + bv;
    }
}

// ---- output: e=ssp(h@ow1+b1); e=ssp(e@ow2+b2); atomic energy[batch] += e@ow3+b3 ----
__global__ __launch_bounds__(32) void k_out(
    const float* __restrict__ h, const int* __restrict__ batch,
    const f16* __restrict__ ow1p, const float* __restrict__ ob1,
    const f16* __restrict__ ow2p, const float* __restrict__ ob2,
    const float* __restrict__ ow3, const float* __restrict__ ob3,
    float* __restrict__ out) {
    __shared__ __align__(16) f16 w[16 * 136];
    const int lane = threadIdx.x;
    const int m = lane & 15, kh = lane >> 4;
    const int n0 = blockIdx.x * 16;
    const float* src = h + (size_t)(n0 + m) * HID;

    v8f acc[8];
    #pragma unroll
    for (int ks = 0; ks < 4; ks++) {
        v16h av;
        #pragma unroll
        for (int q = 0; q < 8; q++) {
            av[q]     = (f16)src[ks * 32 + kh * 8 + q];
            av[q + 8] = (f16)src[ks * 32 + 16 + kh * 8 + q];
        }
        #pragma unroll
        for (int t = 0; t < 8; t++) {
            AB b;
            const uint4* bp = (const uint4*)(ow1p + (size_t)((t * 4 + ks) * 32 + lane) * 16);
            b.u[0] = bp[0]; b.u[1] = bp[1];
            acc[t] = (ks == 0) ? wmma16_z(av, b.v) : wmma16(av, b.v, acc[t]);
        }
    }
    #pragma unroll
    for (int t = 0; t < 8; t++) {
        int col = t * 16 + m;
        float bv = ob1[col];
        #pragma unroll
        for (int r = 0; r < 8; r++)
            w[(r + kh * 8) * 136 + col] = (f16)sspf(acc[t][r] + bv);
    }
    __syncthreads();
    v8f acc2[8];
    #pragma unroll
    for (int ks = 0; ks < 4; ks++) {
        AB a2;
        a2.u[0] = *(const uint4*)&w[m * 136 + ks * 32 + kh * 8];
        a2.u[1] = *(const uint4*)&w[m * 136 + ks * 32 + 16 + kh * 8];
        #pragma unroll
        for (int t = 0; t < 8; t++) {
            AB b;
            const uint4* bp = (const uint4*)(ow2p + (size_t)((t * 4 + ks) * 32 + lane) * 16);
            b.u[0] = bp[0]; b.u[1] = bp[1];
            acc2[t] = (ks == 0) ? wmma16_z(a2.v, b.v) : wmma16(a2.v, b.v, acc2[t]);
        }
    }
    float vr[8];
    #pragma unroll
    for (int r = 0; r < 8; r++) vr[r] = 0.0f;
    #pragma unroll
    for (int t = 0; t < 8; t++) {
        int col = t * 16 + m;
        float w3 = ow3[col];
        float bv = ob2[col];
        #pragma unroll
        for (int r = 0; r < 8; r++) vr[r] += sspf(acc2[t][r] + bv) * w3;
    }
    float b3 = ob3[0];
    #pragma unroll
    for (int r = 0; r < 8; r++) {
        float v = vr[r];
        v += __shfl_xor(v, 1, 32);
        v += __shfl_xor(v, 2, 32);
        v += __shfl_xor(v, 4, 32);
        v += __shfl_xor(v, 8, 32);
        if (m == 0) {
            int node = n0 + r + kh * 8;
            atomicAdd(&out[batch[node]], v + b3);
        }
    }
}

extern "C" void kernel_launch(void* const* d_in, const int* in_sizes, int n_in,
                              void* d_out, int out_size, void* d_ws, size_t ws_size,
                              hipStream_t stream) {
    (void)in_sizes; (void)n_in; (void)out_size; (void)ws_size;
    const int*   z     = (const int*)d_in[0];
    const float* pos   = (const float*)d_in[1];
    const int*   ei    = (const int*)d_in[2];
    const int*   batch = (const int*)d_in[3];
    const float* embed = (const float*)d_in[4];
    const float* fw1 = (const float*)d_in[5];  const float* fb1 = (const float*)d_in[6];
    const float* fw2 = (const float*)d_in[7];  const float* fb2 = (const float*)d_in[8];
    const float* uw1 = (const float*)d_in[9];  const float* ub1 = (const float*)d_in[10];
    const float* uw2 = (const float*)d_in[11]; const float* ub2 = (const float*)d_in[12];
    const float* ow1 = (const float*)d_in[13]; const float* ob1 = (const float*)d_in[14];
    const float* ow2 = (const float*)d_in[15]; const float* ob2 = (const float*)d_in[16];
    const float* ow3 = (const float*)d_in[17]; const float* ob3 = (const float*)d_in[18];

    char* ws = (char*)d_ws;
    size_t off = 0;
    auto take = [&](size_t bytes) -> void* {
        void* p = ws + off;
        off += (bytes + 255) & ~(size_t)255;
        return p;
    };
    float* h    = (float*)take((size_t)NN * HID * 4);   // 25.6 MB
    float* aggr = (float*)take((size_t)NN * HID * 4);   // 25.6 MB
    f16*   rbf  = (f16*)  take((size_t)NE * 32 * 2);    // 51.2 MB
    float* invd = (float*)take((size_t)NN * 4);         // deg then 1/deg
    f16* fw1p = (f16*)take((size_t)NL * 4096 * 2);
    f16* fw2p = (f16*)take((size_t)NL * 16384 * 2);
    f16* uw1p = (f16*)take((size_t)NL * 16384 * 2);
    f16* uw2p = (f16*)take((size_t)NL * 16384 * 2);
    f16* ow1p = (f16*)take(16384 * 2);
    f16* ow2p = (f16*)take(16384 * 2);

    // pack weights into WMMA-B-native layout (f16)
    for (int l = 0; l < NL; l++) {
        k_pack32 <<<16, 256, 0, stream>>>(fw1 + (size_t)l * 4096,  fw1p + (size_t)l * 4096);
        k_pack128<<<64, 256, 0, stream>>>(fw2 + (size_t)l * 16384, fw2p + (size_t)l * 16384);
        k_pack128<<<64, 256, 0, stream>>>(uw1 + (size_t)l * 16384, uw1p + (size_t)l * 16384);
        k_pack128<<<64, 256, 0, stream>>>(uw2 + (size_t)l * 16384, uw2p + (size_t)l * 16384);
    }
    k_pack128<<<64, 256, 0, stream>>>(ow1, ow1p);
    k_pack128<<<64, 256, 0, stream>>>(ow2, ow2p);

    hipMemsetAsync(invd, 0, (size_t)NN * 4, stream);
    k_pre<<<NE / 256, 256, 0, stream>>>(pos, ei, rbf, invd);
    k_invdeg<<<(NN + 255) / 256, 256, 0, stream>>>(invd);
    k_inith<<<(NN * HID) / 256, 256, 0, stream>>>(z, embed, h);

    for (int l = 0; l < NL; l++) {
        hipMemsetAsync(aggr, 0, (size_t)NN * HID * 4, stream);
        k_edge<<<NE / (16 * EW), EW * 32, 0, stream>>>(rbf, ei, h, aggr,
            fw1p + (size_t)l * 4096,  fb1 + (size_t)l * HID,
            fw2p + (size_t)l * 16384, fb2 + (size_t)l * HID);
        k_node<<<NN / 16, 32, 0, stream>>>(aggr, invd, h,
            uw1p + (size_t)l * 16384, ub1 + (size_t)l * HID,
            uw2p + (size_t)l * 16384, ub2 + (size_t)l * HID);
    }

    hipMemsetAsync(d_out, 0, (size_t)NG * 4, stream);
    k_out<<<NN / 16, 32, 0, stream>>>(h, batch, ow1p, ob1, ow2p, ob2, ow3, ob3, (float*)d_out);
}
